// InformerModel_82523501625741
// MI455X (gfx1250) — matrix operand order repaired and checked by reading, not compiled, not run on hardware
//
#include <hip/hip_runtime.h>
#include <math.h>

typedef _Float16 f16;
typedef __attribute__((ext_vector_type(16))) _Float16 v16h;
typedef __attribute__((ext_vector_type(8)))  _Float16 v8h;
typedef __attribute__((ext_vector_type(8)))  float    v8f;

#define BATCH 8
#define SEQ   2048
#define DMODEL 256
#define NHEAD 8
#define DHEAD 32
#define DFF   1024
#define UTOP  40
#define SSAMP 40
#define NROWS (BATCH*SEQ)   // 16384

// ---------------------------------------------------------------------------
// f32 -> f16 conversion (grid-stride)
// ---------------------------------------------------------------------------
__global__ void convert_f32_f16(const float* __restrict__ src,
                                f16* __restrict__ dst, int n) {
  int i = blockIdx.x * blockDim.x + threadIdx.x;
  int stride = gridDim.x * blockDim.x;
  for (; i < n; i += stride) dst[i] = (f16)src[i];
}

// ---------------------------------------------------------------------------
// Transposing f32 -> f16 conversion: dst[n*K + k] = (f16)src[k*N + n]
// 32x32 LDS tile keeps both global sides coalesced. K,N multiples of 32.
// ---------------------------------------------------------------------------
__global__ void transpose_f32_f16(const float* __restrict__ src,
                                  f16* __restrict__ dst, int K, int N) {
  __shared__ float tile[32][33];
  int n0 = blockIdx.x * 32;
  int k0 = blockIdx.y * 32;
  int tx = threadIdx.x & 31;        // 32
  int ty = threadIdx.x >> 5;        // 8 rows of 32
  for (int i = ty; i < 32; i += 8)
    tile[i][tx] = src[(size_t)(k0 + i) * N + n0 + tx];
  __syncthreads();
  for (int i = ty; i < 32; i += 8)
    dst[(size_t)(n0 + i) * K + k0 + tx] = (f16)tile[tx][i];
}

// ---------------------------------------------------------------------------
// Embedding: h = x @ w_in + b_in   (K=32, scalar; writes f32 + f16)
// ---------------------------------------------------------------------------
__global__ void embed_kernel(const float* __restrict__ x,
                             const float* __restrict__ w_in,
                             const float* __restrict__ b_in,
                             float* __restrict__ h32, f16* __restrict__ h16) {
  __shared__ float xr[32];
  int row = blockIdx.x;          // 0..NROWS-1
  int c   = threadIdx.x;         // 0..255
  if (c < 32) xr[c] = x[row * 32 + c];
  __syncthreads();
  float acc = b_in[c];
#pragma unroll
  for (int k = 0; k < 32; ++k) acc += xr[k] * w_in[k * DMODEL + c];
  size_t off = (size_t)row * DMODEL + c;
  h32[off] = acc;
  h16[off] = (f16)acc;
}

// ---------------------------------------------------------------------------
// A-fragment load: 16-bit A 16x32 layout, lane l (m=l&15, half=l>>4):
//   a[e] = A[m][k0 + e + 8*((e>>3)+half)]  -> two contiguous 16B loads.
// ---------------------------------------------------------------------------
__device__ __forceinline__ v16h load_afrag(const f16* __restrict__ base,
                                           int hl) {
  v8h lo = *(const v8h*)(base + 8 * hl);
  v8h hi = *(const v8h*)(base + 16 + 8 * hl);
  v16h a;
#pragma unroll
  for (int e = 0; e < 8; ++e) { a[e] = lo[e]; a[8 + e] = hi[e]; }
  return a;
}

// ---------------------------------------------------------------------------
// WMMA GEMM:  C[M,N] = A_f16[M,K] @ W[K,N] + bias, optional relu,
// where W is supplied PRE-TRANSPOSED as Bt[N,K] (weights are small and
// L2-resident, so redundant cached reads beat LDS transposes + barriers).
//
// Block = 128 threads (4 waves). Block tile 128(M) x 64(N).
// Wave w: rows m0 = blockY*128 + w*32 (two 16-row A fragments),
//         cols n0..n0+63 (four 16-col B fragments) -> 8 accumulators.
// B fragment (32x16): lane l (n=l&15, half=l>>4):
//   b[e] = Bt[n][k0 + e + 8*((e>>3)+half)]  (same k order as A).
// C/D f32: vgpr r -> M = r + 8*half, lane&15 -> N.
// ---------------------------------------------------------------------------
__launch_bounds__(128)
__global__ void gemm_wmma_f16(const f16* __restrict__ A,
                              const f16* __restrict__ Bt,
                              const float* __restrict__ bias,
                              float* __restrict__ Cf, f16* __restrict__ Ch,
                              int M, int N, int K, int relu) {
  const int t    = threadIdx.x;
  const int lane = t & 31;
  const int wave = t >> 5;
  const int hl   = lane >> 4;
  const int lm   = lane & 15;
  const int m0   = blockIdx.y * 128 + wave * 32;
  const int n0   = blockIdx.x * 64;

  v8f acc0[4] = {};
  v8f acc1[4] = {};
  const f16* Arow0 = A + (size_t)(m0 + lm) * K;
  const f16* Arow1 = A + (size_t)(m0 + 16 + lm) * K;

  for (int k0 = 0; k0 < K; k0 += 32) {
    v16h a0 = load_afrag(Arow0 + k0, hl);
    v16h a1 = load_afrag(Arow1 + k0, hl);
#pragma unroll
    for (int j = 0; j < 4; ++j) {
      const f16* bp = Bt + (size_t)(n0 + j * 16 + lm) * K + k0;
      v8h blo = *(const v8h*)(bp + 8 * hl);
      v8h bhi = *(const v8h*)(bp + 16 + 8 * hl);
      v16h b;
#pragma unroll
      for (int e = 0; e < 8; ++e) { b[e] = blo[e]; b[8 + e] = bhi[e]; }
      acc0[j] = __builtin_amdgcn_wmma_f32_16x16x32_f16(
          false, a0, false, b, (short)0, acc0[j], false, false);
      acc1[j] = __builtin_amdgcn_wmma_f32_16x16x32_f16(
          false, a1, false, b, (short)0, acc1[j], false, false);
    }
  }

#pragma unroll
  for (int j = 0; j < 4; ++j) {
    int n = n0 + j * 16 + lm;
    float bv = bias ? bias[n] : 0.f;
#pragma unroll
    for (int r = 0; r < 8; ++r) {
      int m = m0 + r + 8 * hl;
      float v0 = acc0[j][r] + bv;
      float v1 = acc1[j][r] + bv;
      if (relu) {
        v0 = v0 > 0.f ? v0 : 0.f;
        v1 = v1 > 0.f ? v1 : 0.f;
      }
      if (Cf) {
        Cf[(size_t)m * N + n] = v0;
        Cf[(size_t)(m + 16) * N + n] = v1;
      }
      if (Ch) {
        Ch[(size_t)m * N + n] = (f16)v0;
        Ch[(size_t)(m + 16) * N + n] = (f16)v1;
      }
    }
  }
}

// ---------------------------------------------------------------------------
// vmean[b, h*32+d] = mean over L of v[b, l, h*32+d]
// ---------------------------------------------------------------------------
__global__ void vmean_kernel(const float* __restrict__ v,
                             float* __restrict__ vmean) {
  int b = blockIdx.x;
  int c = threadIdx.x;   // 0..255
  float s = 0.f;
  for (int l = 0; l < SEQ; ++l)
    s += v[((size_t)b * SEQ + l) * DMODEL + c];
  vmean[b * DMODEL + c] = s * (1.f / (float)SEQ);
}

// ctx[b,l,:] = vmean[b,:]  (f16, consumed only by the O-projection GEMM)
__global__ void ctx_fill_kernel(const float* __restrict__ vmean,
                                f16* __restrict__ ctx) {
  int row = blockIdx.x;          // b*L + l
  int c   = threadIdx.x;
  int b   = row >> 11;           // /SEQ
  ctx[(size_t)row * DMODEL + c] = (f16)vmean[b * DMODEL + c];
}

// ---------------------------------------------------------------------------
// Sparsity measure: M[b,h,l] = max_s(qk_s) - sum_s(qk_s)/L  (S=40 sampled keys)
// ---------------------------------------------------------------------------
__global__ void measure_kernel(const float* __restrict__ q,
                               const float* __restrict__ k,
                               const int* __restrict__ idx,
                               float* __restrict__ Mout) {
  int gid = blockIdx.x * blockDim.x + threadIdx.x;   // b*H*L + h*L + l
  if (gid >= BATCH * NHEAD * SEQ) return;
  int l = gid & (SEQ - 1);
  int h = (gid >> 11) & (NHEAD - 1);
  int b = gid >> 14;
  const float* qp = q + ((size_t)b * SEQ + l) * DMODEL + h * DHEAD;
  float qr[DHEAD];
#pragma unroll
  for (int d = 0; d < DHEAD; ++d) qr[d] = qp[d];
  float mx = -3.4e38f, sm = 0.f;
  for (int s = 0; s < SSAMP; ++s) {
    int kl = idx[l * SSAMP + s];
    const float* kp = k + ((size_t)b * SEQ + kl) * DMODEL + h * DHEAD;
    float dot = 0.f;
#pragma unroll
    for (int d = 0; d < DHEAD; ++d) dot += qr[d] * kp[d];
    mx = dot > mx ? dot : mx;
    sm += dot;
  }
  Mout[gid] = mx - sm * (1.f / (float)SEQ);
}

// ---------------------------------------------------------------------------
// top-40 over L=2048 per (b,h): iterative parallel argmax with masking.
// ---------------------------------------------------------------------------
__global__ void topk_kernel(const float* __restrict__ Mm,
                            int* __restrict__ top) {
  __shared__ float vals[SEQ];
  __shared__ float rv[256];
  __shared__ int   ri[256];
  int bh = blockIdx.x;     // b*H + h
  int t  = threadIdx.x;    // 0..255
  for (int i = t; i < SEQ; i += 256) vals[i] = Mm[bh * SEQ + i];
  __syncthreads();
  for (int j = 0; j < UTOP; ++j) {
    float bestv = -3.4e38f; int besti = SEQ;
    for (int i = t; i < SEQ; i += 256) {
      float vv = vals[i];
      if (vv > bestv) { bestv = vv; besti = i; }
    }
    rv[t] = bestv; ri[t] = besti;
    __syncthreads();
    for (int s = 128; s > 0; s >>= 1) {
      if (t < s) {
        if (rv[t + s] > rv[t] ||
            (rv[t + s] == rv[t] && ri[t + s] < ri[t])) {
          rv[t] = rv[t + s]; ri[t] = ri[t + s];
        }
      }
      __syncthreads();
    }
    if (t == 0) { top[bh * UTOP + j] = ri[0]; vals[ri[0]] = -3.4e38f; }
    __syncthreads();
  }
}

// ---------------------------------------------------------------------------
// For each selected query row: scores over all L keys, softmax, attn@V,
// scatter into ctx (f16) at its sequence position.
// ---------------------------------------------------------------------------
__global__ void attn_topq_kernel(const float* __restrict__ q,
                                 const float* __restrict__ k,
                                 const float* __restrict__ v,
                                 const int* __restrict__ top,
                                 f16* __restrict__ ctx) {
  __shared__ float sc[SEQ];
  __shared__ float qs[DHEAD];
  __shared__ float red[256];
  __shared__ float updp[256];
  int blk = blockIdx.x;               // b*(H*U) + h*U + j
  int j = blk % UTOP;
  int h = (blk / UTOP) & (NHEAD - 1);
  int b = blk / (UTOP * NHEAD);
  int t = threadIdx.x;
  int bh = b * NHEAD + h;
  int tl = top[bh * UTOP + j];
  if (t < DHEAD) qs[t] = q[((size_t)b * SEQ + tl) * DMODEL + h * DHEAD + t];
  __syncthreads();
  const float scale = 0.17677669529663687f;   // 1/sqrt(32)
  for (int l = t; l < SEQ; l += 256) {
    const float* kp = k + ((size_t)b * SEQ + l) * DMODEL + h * DHEAD;
    float dot = 0.f;
#pragma unroll
    for (int d = 0; d < DHEAD; ++d) dot += qs[d] * kp[d];
    sc[l] = dot * scale;
  }
  __syncthreads();
  float mx = -3.4e38f;
  for (int l = t; l < SEQ; l += 256) mx = sc[l] > mx ? sc[l] : mx;
  red[t] = mx; __syncthreads();
  for (int s = 128; s > 0; s >>= 1) {
    if (t < s && red[t + s] > red[t]) red[t] = red[t + s];
    __syncthreads();
  }
  mx = red[0]; __syncthreads();
  float sm = 0.f;
  for (int l = t; l < SEQ; l += 256) {
    float e = expf(sc[l] - mx);
    sc[l] = e; sm += e;
  }
  red[t] = sm; __syncthreads();
  for (int s = 128; s > 0; s >>= 1) {
    if (t < s) red[t] += red[t + s];
    __syncthreads();
  }
  float inv = 1.f / red[0];
  int d = t & (DHEAD - 1), slice = t >> 5;   // 8 slices of 256 keys
  float acc = 0.f;
  for (int l = slice * 256; l < slice * 256 + 256; ++l)
    acc += sc[l] * v[((size_t)b * SEQ + l) * DMODEL + h * DHEAD + d];
  updp[t] = acc; __syncthreads();
  if (t < DHEAD) {
    float s3 = 0.f;
#pragma unroll
    for (int sl = 0; sl < 8; ++sl) s3 += updp[sl * 32 + t];
    ctx[((size_t)b * SEQ + tl) * DMODEL + h * DHEAD + t] = (f16)(s3 * inv);
  }
}

// ---------------------------------------------------------------------------
// h = LayerNorm(h + addv) * g + beta ; writes f32 (in place) + f16 shadow.
// ---------------------------------------------------------------------------
__global__ void add_ln_kernel(float* __restrict__ h32, f16* __restrict__ h16,
                              const float* __restrict__ addv,
                              const float* __restrict__ g,
                              const float* __restrict__ beta) {
  __shared__ float red[256];
  int row = blockIdx.x;
  int c   = threadIdx.x;
  size_t off = (size_t)row * DMODEL + c;
  float xv = h32[off] + addv[off];
  red[c] = xv; __syncthreads();
  for (int s = 128; s > 0; s >>= 1) {
    if (c < s) red[c] += red[c + s];
    __syncthreads();
  }
  float mean = red[0] * (1.f / (float)DMODEL);
  __syncthreads();
  float dv = xv - mean;
  red[c] = dv * dv; __syncthreads();
  for (int s = 128; s > 0; s >>= 1) {
    if (c < s) red[c] += red[c + s];
    __syncthreads();
  }
  float var = red[0] * (1.f / (float)DMODEL);
  float out = dv * rsqrtf(var + 1e-5f) * g[c] + beta[c];
  h32[off] = out;
  h16[off] = (f16)out;
}

// ---------------------------------------------------------------------------
// Head: out[b] = relu(h[b,L-1,:] @ w_fc1 + b_fc1) @ w_fc2 + b_fc2
// ---------------------------------------------------------------------------
__global__ void head_kernel(const float* __restrict__ h32,
                            const float* __restrict__ w1,
                            const float* __restrict__ b1,
                            const float* __restrict__ w2,
                            const float* __restrict__ b2,
                            float* __restrict__ outp) {
  __shared__ float t1[64];
  int b = blockIdx.x;
  int t = threadIdx.x;   // 0..63
  const float* last = h32 + ((size_t)b * SEQ + (SEQ - 1)) * DMODEL;
  float acc = b1[t];
  for (int kk = 0; kk < DMODEL; ++kk) acc += last[kk] * w1[kk * 64 + t];
  t1[t] = acc > 0.f ? acc : 0.f;
  __syncthreads();
  if (t == 0) {
    float o = b2[0];
    for (int kk = 0; kk < 64; ++kk) o += t1[kk] * w2[kk];
    outp[b] = o;
  }
}

// ---------------------------------------------------------------------------
static inline void launch_gemm(const f16* A, const f16* Bt, const float* bias,
                               float* Cf, f16* Ch, int M, int N, int K,
                               int relu, hipStream_t stream) {
  dim3 grid(N / 64, M / 128);
  gemm_wmma_f16<<<grid, 128, 0, stream>>>(A, Bt, bias, Cf, Ch, M, N, K, relu);
}

extern "C" void kernel_launch(void* const* d_in, const int* in_sizes, int n_in,
                              void* d_out, int out_size, void* d_ws,
                              size_t ws_size, hipStream_t stream) {
  (void)in_sizes; (void)n_in; (void)out_size; (void)ws_size;
  const float* x      = (const float*)d_in[0];
  const int*   idxs   = (const int*)  d_in[1];
  const float* w_in   = (const float*)d_in[2];
  const float* b_in   = (const float*)d_in[3];
  const float* wq     = (const float*)d_in[4];
  const float* bq     = (const float*)d_in[5];
  const float* wk     = (const float*)d_in[6];
  const float* bk     = (const float*)d_in[7];
  const float* wv     = (const float*)d_in[8];
  const float* bv     = (const float*)d_in[9];
  const float* wo     = (const float*)d_in[10];
  const float* bo     = (const float*)d_in[11];
  const float* w_ff1  = (const float*)d_in[12];
  const float* b_ff1  = (const float*)d_in[13];
  const float* w_ff2  = (const float*)d_in[14];
  const float* b_ff2  = (const float*)d_in[15];
  const float* ln1_g  = (const float*)d_in[16];
  const float* ln1_b  = (const float*)d_in[17];
  const float* ln2_g  = (const float*)d_in[18];
  const float* ln2_b  = (const float*)d_in[19];
  const float* w_fc1  = (const float*)d_in[20];
  const float* b_fc1  = (const float*)d_in[21];
  const float* w_fc2  = (const float*)d_in[22];
  const float* b_fc2  = (const float*)d_in[23];

  // ---- workspace carve (256B aligned) ----
  char* p = (char*)d_ws;
  auto carve = [&](size_t bytes) -> void* {
    void* r = (void*)p;
    p += (bytes + 255) & ~(size_t)255;
    return r;
  };
  float* h32   = (float*)carve((size_t)NROWS * DMODEL * 4);
  f16*   h16   = (f16*)  carve((size_t)NROWS * DMODEL * 2);
  float* q32   = (float*)carve((size_t)NROWS * DMODEL * 4);
  float* k32   = (float*)carve((size_t)NROWS * DMODEL * 4);
  float* v32   = (float*)carve((size_t)NROWS * DMODEL * 4);
  float* tmp32 = (float*)carve((size_t)NROWS * DMODEL * 4);
  f16*   ctx16 = (f16*)  carve((size_t)NROWS * DMODEL * 2);
  f16*   y116  = (f16*)  carve((size_t)NROWS * DFF * 2);
  // transposed (N-major) f16 weights
  f16*   wq16  = (f16*)  carve((size_t)2 * DMODEL * DMODEL * 2);
  f16*   wk16  = (f16*)  carve((size_t)2 * DMODEL * DMODEL * 2);
  f16*   wv16  = (f16*)  carve((size_t)2 * DMODEL * DMODEL * 2);
  f16*   wo16  = (f16*)  carve((size_t)2 * DMODEL * DMODEL * 2);
  f16*   wf116 = (f16*)  carve((size_t)2 * DMODEL * DFF * 2);
  f16*   wf216 = (f16*)  carve((size_t)2 * DFF * DMODEL * 2);
  float* vmean = (float*)carve((size_t)BATCH * DMODEL * 4);
  float* Mm    = (float*)carve((size_t)BATCH * NHEAD * SEQ * 4);
  int*   top   = (int*)  carve((size_t)BATCH * NHEAD * UTOP * 4);

  // ---- convert + transpose weights to N-major f16, once per layer ----
  auto tconv = [&](const float* s, f16* d, int K, int N) {
    for (int i = 0; i < 2; ++i) {
      dim3 grid(N / 32, K / 32);
      transpose_f32_f16<<<grid, 256, 0, stream>>>(
          s + (size_t)i * K * N, d + (size_t)i * K * N, K, N);
    }
  };
  tconv(wq, wq16, DMODEL, DMODEL);
  tconv(wk, wk16, DMODEL, DMODEL);
  tconv(wv, wv16, DMODEL, DMODEL);
  tconv(wo, wo16, DMODEL, DMODEL);
  tconv(w_ff1, wf116, DMODEL, DFF);
  tconv(w_ff2, wf216, DFF, DMODEL);

  // ---- embedding ----
  embed_kernel<<<NROWS, DMODEL, 0, stream>>>(x, w_in, b_in, h32, h16);

  for (int i = 0; i < 2; ++i) {
    const f16* wqi  = wq16  + (size_t)i * DMODEL * DMODEL;
    const f16* wki  = wk16  + (size_t)i * DMODEL * DMODEL;
    const f16* wvi  = wv16  + (size_t)i * DMODEL * DMODEL;
    const f16* woi  = wo16  + (size_t)i * DMODEL * DMODEL;
    const f16* wf1i = wf116 + (size_t)i * DMODEL * DFF;
    const f16* wf2i = wf216 + (size_t)i * DFF * DMODEL;

    // Q,K,V projections (WMMA)
    launch_gemm(h16, wqi, bq + i * DMODEL, q32, nullptr, NROWS, DMODEL, DMODEL, 0, stream);
    launch_gemm(h16, wki, bk + i * DMODEL, k32, nullptr, NROWS, DMODEL, DMODEL, 0, stream);
    launch_gemm(h16, wvi, bv + i * DMODEL, v32, nullptr, NROWS, DMODEL, DMODEL, 0, stream);

    // prob-sparse attention (scalar, memory-bound)
    vmean_kernel<<<BATCH, DMODEL, 0, stream>>>(v32, vmean);
    ctx_fill_kernel<<<NROWS, DMODEL, 0, stream>>>(vmean, ctx16);
    measure_kernel<<<(BATCH * NHEAD * SEQ + 255) / 256, 256, 0, stream>>>(
        q32, k32, idxs + (size_t)i * SEQ * SSAMP, Mm);
    topk_kernel<<<BATCH * NHEAD, 256, 0, stream>>>(Mm, top);
    attn_topq_kernel<<<BATCH * NHEAD * UTOP, 256, 0, stream>>>(
        q32, k32, v32, top, ctx16);

    // output projection + residual LN1
    launch_gemm(ctx16, woi, bo + i * DMODEL, tmp32, nullptr, NROWS, DMODEL, DMODEL, 0, stream);
    add_ln_kernel<<<NROWS, DMODEL, 0, stream>>>(h32, h16, tmp32,
                                                ln1_g + i * DMODEL, ln1_b + i * DMODEL);

    // FFN (WMMA x2) + residual LN2
    launch_gemm(h16, wf1i, b_ff1 + i * DFF, nullptr, y116, NROWS, DFF, DMODEL, 1, stream);
    launch_gemm(y116, wf2i, b_ff2 + i * DMODEL, tmp32, nullptr, NROWS, DMODEL, DFF, 0, stream);
    add_ln_kernel<<<NROWS, DMODEL, 0, stream>>>(h32, h16, tmp32,
                                                ln2_g + i * DMODEL, ln2_b + i * DMODEL);
  }

  head_kernel<<<BATCH, 64, 0, stream>>>(h32, w_fc1, b_fc1, w_fc2, b_fc2,
                                        (float*)d_out);
}
